// BlockLSTM_6665789243437
// MI455X (gfx1250) — compile-verified, matrix operand order checked by reading, not run on hardware
//
#include <hip/hip_runtime.h>
#include <hip/hip_bf16.h>

#define BB 64
#define SS 1024
#define INN 1024
#define HH 1024

typedef __attribute__((ext_vector_type(16))) __bf16 v16bf;
typedef __attribute__((ext_vector_type(8)))  __bf16 v8bf;
typedef __attribute__((ext_vector_type(8)))  float  v8f;

__device__ __forceinline__ float sigmoidf_(float v) {
    return 1.0f / (1.0f + __expf(-v));
}

// Stable fast tanh: sign(x) * (1 - e^(-2|x|)) / (1 + e^(-2|x|)); one v_exp + rcp.
__device__ __forceinline__ float tanhf_fast(float v) {
    float a = __builtin_fabsf(v);
    float e = __expf(-2.0f * a);
    float r = (1.0f - e) / (1.0f + e);
    return __builtin_copysignf(r, v);
}

// A-operand fragment (16x32 bf16) from an fp32 row-major row.
// Lane holds row M = lane%16; K elements [k0, k0+8) in e0..7, [k0+16, k0+24) in e8..15.
__device__ __forceinline__ v16bf load_a_f32(const float* __restrict__ row, int k0) {
    v16bf a;
    const float* p0 = row + k0;
    const float* p1 = row + k0 + 16;
#pragma unroll
    for (int e = 0; e < 8; ++e) {
        a[e]     = (__bf16)p0[e];
        a[e + 8] = (__bf16)p1[e];
    }
    return a;
}

// B-operand fragment (32x16 bf16) from pre-transposed bf16 weights ([N][K] row-major):
// lane holds column N = lane%16, same K-chunk pattern -> two contiguous 16B loads.
__device__ __forceinline__ v16bf load_b_bf16(const __bf16* __restrict__ row, int k0) {
    v8bf lo = *(const v8bf*)(row + k0);
    v8bf hi = *(const v8bf*)(row + k0 + 16);
    v16bf b;
#pragma unroll
    for (int e = 0; e < 8; ++e) { b[e] = lo[e]; b[e + 8] = hi[e]; }
    return b;
}

// One-time per call: transpose + downconvert weights.
// WxT[k][h][i] = (bf16) Wx[k][i][h]  (and same for Wh).
__global__ void prep_weights_kernel(const float* __restrict__ Wx,
                                    const float* __restrict__ Wh,
                                    __bf16* __restrict__ WxT,
                                    __bf16* __restrict__ WhT) {
    size_t idx = (size_t)blockIdx.x * blockDim.x + threadIdx.x;   // < 4*HH*INN
    int    i   = (int)(idx & (INN - 1));
    size_t r   = idx >> 10;
    int    h   = (int)(r & (HH - 1));
    int    k   = (int)(r >> 10);
    size_t src = (size_t)k * INN * HH + (size_t)i * HH + h;
    WxT[idx] = (__bf16)Wx[src];
    WhT[idx] = (__bf16)Wh[src];
}

// One LSTM time step: g = x_t*Wx + h*Wh + bx + bh (bf16 WMMA, f32 acc),
// gates + state update fused. Each wave: one 16(b) x 16(h) patch, all 4 gates.
// Block = 4 waves = the 4 M-tiles of one H-tile (B fragments shared via WGP$).
// Carry swap per reference: h_out <- c_new, c_out <- h_new.
__global__ __launch_bounds__(128, 1)
void lstm_step_kernel(const float* __restrict__ x, int t,
                      const __bf16* __restrict__ WxT,
                      const __bf16* __restrict__ WhT,
                      const float* __restrict__ bx,
                      const float* __restrict__ bh,
                      const float* __restrict__ h_in,
                      const float* __restrict__ c_in,
                      float* __restrict__ h_out,
                      float* __restrict__ c_out,
                      float* __restrict__ out,
                      float* __restrict__ fin) {
    const int lane  = threadIdx.x & 31;
    const int wave  = threadIdx.x >> 5;
    const int gid   = blockIdx.x * 4 + wave;   // 0..255
    const int mt    = gid & 3;                 // 4 tiles over B=64
    const int ht    = gid >> 2;                // 64 tiles over H=1024
    const int lmod  = lane & 15;
    const int lhalf = lane >> 4;
    const int k0    = lhalf * 8;

    const int hcol = ht * 16 + lmod;           // this lane's output column
    const int arow = mt * 16 + lmod;           // this lane's A-operand row

    // Accumulators initialized with the (column-only) biases.
    v8f acc[4];
#pragma unroll
    for (int k = 0; k < 4; ++k) {
        float bias = bx[k * HH + hcol] + bh[k * HH + hcol];
#pragma unroll
        for (int r = 0; r < 8; ++r) acc[k][r] = bias;
    }

    const __bf16* wxr = WxT + (size_t)hcol * INN;   // gate k at +k*HH*INN
    const __bf16* whr = WhT + (size_t)hcol * INN;

    // --- x_t @ Wx ---
    const float* xrow = x + (size_t)arow * (SS * INN) + (size_t)t * INN;
    for (int kb = 0; kb < INN; kb += 32) {
        v16bf a = load_a_f32(xrow, kb + k0);
#pragma unroll
        for (int k = 0; k < 4; ++k) {
            v16bf bf = load_b_bf16(wxr + (size_t)k * (HH * INN), kb + k0);
            acc[k] = __builtin_amdgcn_wmma_f32_16x16x32_bf16(
                false, a, false, bf, (short)0, acc[k], false, false);
        }
    }

    // --- h @ Wh ---
    const float* hrow = h_in + (size_t)arow * HH;
    for (int kb = 0; kb < HH; kb += 32) {
        v16bf a = load_a_f32(hrow, kb + k0);
#pragma unroll
        for (int k = 0; k < 4; ++k) {
            v16bf bf = load_b_bf16(whr + (size_t)k * (HH * INN), kb + k0);
            acc[k] = __builtin_amdgcn_wmma_f32_16x16x32_bf16(
                false, a, false, bf, (short)0, acc[k], false, false);
        }
    }

    // --- gates + state update (C/D layout: vgpr r -> row M=r + 8*lhalf, col N=lmod) ---
#pragma unroll
    for (int r = 0; r < 8; ++r) {
        int   brow = mt * 16 + lhalf * 8 + r;
        float iv = sigmoidf_(acc[0][r]);
        float fv = sigmoidf_(acc[1][r]);
        float gv = tanhf_fast(acc[2][r]);
        float ov = sigmoidf_(acc[3][r]);
        float cold = c_in[brow * HH + hcol];
        float cnew = fv * cold + iv * gv;
        float hnew = ov * tanhf_fast(cnew);
        out[(size_t)brow * (SS * HH) + (size_t)t * HH + hcol] = ov;
        h_out[brow * HH + hcol] = cnew;   // swapped carry
        c_out[brow * HH + hcol] = hnew;
        if (fin) {                        // last step: h_fin = c_new, c_fin = h_new
            fin[brow * HH + hcol]           = cnew;
            fin[BB * HH + brow * HH + hcol] = hnew;
        }
    }
}

extern "C" void kernel_launch(void* const* d_in, const int* in_sizes, int n_in,
                              void* d_out, int out_size, void* d_ws, size_t ws_size,
                              hipStream_t stream) {
    const float* x  = (const float*)d_in[0];
    const float* h0 = (const float*)d_in[1];
    const float* c0 = (const float*)d_in[2];
    const float* Wx = (const float*)d_in[3];
    const float* bx = (const float*)d_in[4];
    const float* Wh = (const float*)d_in[5];
    const float* bh = (const float*)d_in[6];

    float* out = (float*)d_out;
    float* fin = out + (size_t)BB * SS * HH;

    // Workspace layout: bf16 WxT (8MB) | bf16 WhT (8MB) | 4 x [B,H] f32 states (1MB)
    __bf16* WxT = (__bf16*)d_ws;
    __bf16* WhT = WxT + (size_t)4 * HH * INN;
    float*  st  = (float*)(WhT + (size_t)4 * HH * INN);
    float* hA = st;
    float* cA = hA + BB * HH;
    float* hB = cA + BB * HH;
    float* cB = hB + BB * HH;

    prep_weights_kernel<<<(4 * HH * INN) / 256, 256, 0, stream>>>(Wx, Wh, WxT, WhT);

    for (int t = 0; t < SS; ++t) {
        const float *hi, *ci;
        float *ho, *co;
        if (t == 0)                  { hi = h0; ci = c0; }
        else if (((t - 1) & 1) == 0) { hi = hA; ci = cA; }
        else                         { hi = hB; ci = cB; }
        if ((t & 1) == 0) { ho = hA; co = cA; }
        else              { ho = hB; co = cB; }

        lstm_step_kernel<<<64, 128, 0, stream>>>(
            x, t, WxT, WhT, bx, bh, hi, ci, ho, co,
            out, (t == SS - 1) ? fin : nullptr);
    }
}
